// ConvCrossAttention_52132313039214
// MI455X (gfx1250) — matrix operand order repaired and checked by reading, not compiled, not run on hardware
//
#include <hip/hip_runtime.h>
#include <stdint.h>

typedef _Float16 f16;
typedef __attribute__((ext_vector_type(16))) _Float16 v16h;
typedef __attribute__((ext_vector_type(8)))  float    v8f;
typedef __attribute__((ext_vector_type(4)))  float    v4f;
typedef __attribute__((ext_vector_type(4)))  uint32_t u32x4;

__device__ __forceinline__ u32x4 ld128(const f16* p) { return *(const u32x4*)p; }

// ---------------------------------------------------------------------------
__global__ void cast_f16_kernel(const float* __restrict__ in, f16* __restrict__ out,
                                size_t n) {
    size_t i = (size_t)blockIdx.x * blockDim.x + threadIdx.x;
    if (i < n) out[i] = (f16)in[i];
}

__global__ void zero_f16_kernel(f16* __restrict__ p, size_t n) {
    size_t i = (size_t)blockIdx.x * blockDim.x + threadIdx.x;
    if (i < n) p[i] = (f16)0.f;
}

// ---------------------------------------------------------------------------
// Repack conv weights f32 [Cout][512][3][3] -> f16 [9][Cout][512] (tap-major,
// K contiguous per row so WMMA A-fragments are b128 loads).
__global__ void pack_w_kernel(const float* __restrict__ w, f16* __restrict__ wp,
                              int Cout, size_t total) {
    size_t idx = (size_t)blockIdx.x * blockDim.x + threadIdx.x;
    if (idx >= total) return;               // total = 9*Cout*512
    int c  = (int)(idx % 512);
    int co = (int)((idx / 512) % Cout);
    int j  = (int)(idx / (512ull * Cout));
    wp[idx] = (f16)w[(size_t)co * 4608 + (size_t)c * 9 + j];
}

// ---------------------------------------------------------------------------
// Zero-pad + transpose input: f32 [B][512][H][W] -> f16 [B][H+2][W+2][512]
// pixel-major. Removes all boundary checks from the conv hot loop.
__global__ void pad_in_kernel(const float* __restrict__ x, f16* __restrict__ xp,
                              int H, int W, size_t total) {
    size_t idx = (size_t)blockIdx.x * blockDim.x + threadIdx.x;
    if (idx >= total) return;               // total = B*(H+2)*(W+2)*512
    int c  = (int)(idx % 512); size_t t = idx / 512;
    int xx = (int)(t % (W + 2)); t /= (W + 2);
    int yy = (int)(t % (H + 2)); int b = (int)(t / (H + 2));
    f16 v = (f16)0.f;
    if (yy >= 1 && yy <= H && xx >= 1 && xx <= W)
        v = (f16)x[(((size_t)b * 512 + c) * H + (yy - 1)) * W + (xx - 1)];
    xp[idx] = v;
}

// ---------------------------------------------------------------------------
// RoPE fused with f32->f16 cast, pixel-major [B][S][heads*128].
// Thread order: pair index i fastest -> coalesced reads and writes.
__global__ void rope_cast_kernel(const float* __restrict__ qf, f16* __restrict__ qp,
                                 int heads, int S, size_t total) {
    size_t idx = (size_t)blockIdx.x * blockDim.x + threadIdx.x;
    if (idx >= total) return;               // total = B*heads*64*S
    int i = (int)(idx % 64); size_t t = idx / 64;
    int s = (int)(t % S); t /= S;
    int h = (int)(t % heads); int b = (int)(t / heads);
    size_t base = ((size_t)b * S + s) * (heads * 128) + h * 128 + 2 * i;
    float x0 = qf[base], x1 = qf[base + 1];
    float theta = __powf(10000.0f, -(float)(2 * i) * (1.0f / 128.0f));
    float sn, cs; __sincosf((float)s * theta, &sn, &cs);
    qp[base]     = (f16)(x0 * cs - x1 * sn);
    qp[base + 1] = (f16)(x0 * sn + x1 * cs);
}

// ---------------------------------------------------------------------------
// V: pixel-major f32 [B][Sk][256] -> channel-major f16 [B][256][Sk]
// (V's WMMA K-dim is the pixel axis, so channel-major gives b128 fragments).
__global__ void vtrans_kernel(const float* __restrict__ vf, f16* __restrict__ vh,
                              size_t total) {
    size_t idx = (size_t)blockIdx.x * blockDim.x + threadIdx.x;
    if (idx >= total) return;               // total = B*256*1024, s fastest
    int s = (int)(idx % 1024); size_t t = idx / 1024;
    int c = (int)(t % 256); int b = (int)(t / 256);
    vh[idx] = (f16)vf[((size_t)b * 1024 + s) * 256 + c];
}

// ---------------------------------------------------------------------------
__global__ void lam_kernel(const float* __restrict__ q1, const float* __restrict__ q2,
                           const float* __restrict__ k1, const float* __restrict__ k2,
                           float* __restrict__ lam) {
    int h = threadIdx.x;
    if (h >= 8) return;
    float d1 = 0.f, d2 = 0.f;
    for (int d = 0; d < 64; ++d) {
        d1 += q1[h * 64 + d] * k1[h * 64 + d];
        d2 += q2[h * 64 + d] * k2[h * 64 + d];
    }
    lam[h] = -(__expf(d1) - __expf(d2) + 0.2f);   // sign (-1)^1 folded in
}

// ---------------------------------------------------------------------------
// 3x3 SAME conv as 9 shifted GEMMs, all-WMMA, branch-free hot loop.
// xpad: [B][H+2][W+2][512] f16, wp: [9][Cout][512] f16.
// out : pixel-major [B][S][Cout] f32 (chan_major=0) or [B][Cout][S] (=1).
// Block 128 threads (4 waves); block tile M=32 x N=64.
__global__ void conv3x3_wmma_kernel(const f16* __restrict__ xpad,
                                    const f16* __restrict__ wp,
                                    float* __restrict__ out,
                                    int Cout, int H, int W, int chan_major) {
    const int S = H * W;
    const int Wp = W + 2;
    const int m0 = blockIdx.x * 32;
    const int s0 = blockIdx.y * 64;
    const int b  = blockIdx.z;
    const int lane = threadIdx.x & 31;
    const int wave = threadIdx.x >> 5;
    const int half = lane >> 4;
    const int n    = lane & 15;
    const int m    = lane & 15;

    const int s  = s0 + wave * 16 + n;
    const int py = s / W, px = s % W;
    const f16* inb = xpad + (size_t)b * (H + 2) * Wp * 512;

    v8f acc0 = {}, acc1 = {};
#pragma unroll
    for (int j = 0; j < 9; ++j) {
        const int jy = j / 3, jx = j % 3;
        const f16* bbase  = inb + ((size_t)(py + jy) * Wp + (px + jx)) * 512;
        const f16* a0base = wp + ((size_t)j * Cout + (m0 + m)) * 512;
        const f16* a1base = a0base + 16 * 512;
        if (j < 8) {
            const int ny = (j + 1) / 3, nx = (j + 1) % 3;
            __builtin_prefetch(inb + ((size_t)(py + ny) * Wp + (px + nx)) * 512, 0, 1);
        }
        for (int c0 = 0; c0 < 512; c0 += 32) {
            v16h bfrag, af0, af1;
            ((u32x4*)&bfrag)[0] = ld128(bbase  + c0 + 8 * half);
            ((u32x4*)&bfrag)[1] = ld128(bbase  + c0 + 16 + 8 * half);
            ((u32x4*)&af0)[0]   = ld128(a0base + c0 + 8 * half);
            ((u32x4*)&af0)[1]   = ld128(a0base + c0 + 16 + 8 * half);
            ((u32x4*)&af1)[0]   = ld128(a1base + c0 + 8 * half);
            ((u32x4*)&af1)[1]   = ld128(a1base + c0 + 16 + 8 * half);
            acc0 = __builtin_amdgcn_wmma_f32_16x16x32_f16(
                false, af0, false, bfrag, (short)0, acc0, false, false);
            acc1 = __builtin_amdgcn_wmma_f32_16x16x32_f16(
                false, af1, false, bfrag, (short)0, acc1, false, false);
        }
    }

    if (chan_major) {
        float* ob = out + (size_t)b * Cout * S;
#pragma unroll
        for (int r = 0; r < 8; ++r) {
            ob[(size_t)(m0 + r + 8 * half) * S + s]      = acc0[r];
            ob[(size_t)(m0 + 16 + r + 8 * half) * S + s] = acc1[r];
        }
    } else {
        // C-layout rows are consecutive channels -> vector stores
        float* ob = out + ((size_t)b * S + s) * Cout + m0;
        *(v4f*)(ob + 8 * half)          = ((v4f*)&acc0)[0];
        *(v4f*)(ob + 8 * half + 4)      = ((v4f*)&acc0)[1];
        *(v4f*)(ob + 16 + 8 * half)     = ((v4f*)&acc1)[0];
        *(v4f*)(ob + 16 + 8 * half + 4) = ((v4f*)&acc1)[1];
    }
}

// ---------------------------------------------------------------------------
// Differential flash attention, all operands b128, P redistributed via LDS.
// qp: [B][4096][1024] f16 pixel-major, kp: [B][1024][512] f16 pixel-major,
// vh: [B][256][1024] f16 channel-major.
// Output written into padded pixel-major buffer apad [B][66][66][512] f16.
__global__ void diff_attn_wmma_kernel(const f16* __restrict__ qp,
                                      const f16* __restrict__ kp,
                                      const f16* __restrict__ vh,
                                      const float* __restrict__ lam,
                                      f16* __restrict__ apad) {
    const int Sq = 4096, Sk = 1024;
    const int h    = blockIdx.y;
    const int b    = blockIdx.z;
    const int hkv  = h >> 1;
    const int lane = threadIdx.x & 31;
    const int wave = threadIdx.x >> 5;
    const int half = lane >> 4;
    const int n    = lane & 15;
    const int m    = lane & 15;
    const int q0   = blockIdx.x * 64 + wave * 16;

    __shared__ f16 Pbuf[4][2][16 * 32];

    const f16* vb = vh + (size_t)b * 256 * Sk;

    // Q A-fragments [component][32-wide K slice] : 2x b128 each
    const f16* qrow = qp + ((size_t)b * Sq + q0 + m) * 1024 + h * 128;
    v16h qA[2][2];
#pragma unroll
    for (int c = 0; c < 2; ++c)
#pragma unroll
        for (int ksl = 0; ksl < 2; ++ksl) {
            const f16* p = qrow + c * 64 + ksl * 32;
            ((u32x4*)&qA[c][ksl])[0] = ld128(p + 8 * half);
            ((u32x4*)&qA[c][ksl])[1] = ld128(p + 16 + 8 * half);
        }

    v8f zero = {};
    v8f O[2][4];
    float mx[2][8], sm[2][8];
#pragma unroll
    for (int c = 0; c < 2; ++c) {
#pragma unroll
        for (int d = 0; d < 4; ++d) O[c][d] = zero;
#pragma unroll
        for (int r = 0; r < 8; ++r) { mx[c][r] = -1e30f; sm[c][r] = 0.f; }
    }
    const float scale = 0.125f;

    for (int kv0 = 0; kv0 < Sk; kv0 += 32) {
        if (kv0 + 32 < Sk)
            __builtin_prefetch(kp + ((size_t)b * Sk + kv0 + 32 + n) * 512 + hkv * 128, 0, 1);
        // --- scores ---
        v8f sc[2][2];
#pragma unroll
        for (int t = 0; t < 2; ++t) {
            const f16* krow = kp + ((size_t)b * Sk + kv0 + t * 16 + n) * 512 + hkv * 128;
#pragma unroll
            for (int c = 0; c < 2; ++c) {
                v8f sv = zero;
#pragma unroll
                for (int ksl = 0; ksl < 2; ++ksl) {
                    v16h kf;
                    const f16* p = krow + c * 64 + ksl * 32;
                    ((u32x4*)&kf)[0] = ld128(p + 8 * half);
                    ((u32x4*)&kf)[1] = ld128(p + 16 + 8 * half);
                    sv = __builtin_amdgcn_wmma_f32_16x16x32_f16(
                        false, qA[c][ksl], false, kf, (short)0, sv, false, false);
                }
                sc[c][t] = sv;
            }
        }
        // --- online softmax (both components), P -> LDS as f16 ---
#pragma unroll
        for (int c = 0; c < 2; ++c)
#pragma unroll
            for (int r = 0; r < 8; ++r) {
                float a0 = sc[c][0][r] * scale;
                float a1 = sc[c][1][r] * scale;
                float tm = fmaxf(a0, a1);
#pragma unroll
                for (int msk = 1; msk < 16; msk <<= 1)
                    tm = fmaxf(tm, __shfl_xor(tm, msk, 32));
                float newm  = fmaxf(mx[c][r], tm);
                float alpha = __expf(mx[c][r] - newm);
                mx[c][r] = newm;
                float p0 = __expf(a0 - newm);
                float p1 = __expf(a1 - newm);
                float ts = p0 + p1;
#pragma unroll
                for (int msk = 1; msk < 16; msk <<= 1)
                    ts += __shfl_xor(ts, msk, 32);
                sm[c][r] = sm[c][r] * alpha + ts;
#pragma unroll
                for (int d = 0; d < 4; ++d) O[c][d][r] *= alpha;
                int row = r + 8 * half;
                Pbuf[wave][c][row * 32 + n]      = (f16)p0;
                Pbuf[wave][c][row * 32 + 16 + n] = (f16)p1;
            }
        __syncthreads();
        // --- V B-fragments: kv contiguous -> b128 ---
        v16h vfr[4];
#pragma unroll
        for (int d = 0; d < 4; ++d) {
            const f16* p = vb + (size_t)(hkv * 64 + d * 16 + n) * Sk + kv0 + 8 * half;
            ((u32x4*)&vfr[d])[0] = ld128(p);
            ((u32x4*)&vfr[d])[1] = ld128(p + 16);
        }
        // --- P x V ---
#pragma unroll
        for (int c = 0; c < 2; ++c) {
            v16h pf;
            const uint32_t* Pu = (const uint32_t*)Pbuf[wave][c];
            ((u32x4*)&pf)[0] = *(const u32x4*)(Pu + m * 16 + 4 * half);
            ((u32x4*)&pf)[1] = *(const u32x4*)(Pu + m * 16 + 8 + 4 * half);
#pragma unroll
            for (int d = 0; d < 4; ++d)
                O[c][d] = __builtin_amdgcn_wmma_f32_16x16x32_f16(
                    false, pf, false, vfr[d], (short)0, O[c][d], false, false);
        }
        __syncthreads();
    }

    // --- combine, per-row RMSNorm, store into padded pixel-major buffer ---
    const float lamh = lam[h];
#pragma unroll
    for (int r = 0; r < 8; ++r) {
        float inv0 = 1.0f / sm[0][r];
        float inv1 = lamh / sm[1][r];
        float vals[4], ss = 0.f;
#pragma unroll
        for (int d = 0; d < 4; ++d) {
            float o = O[0][d][r] * inv0 + O[1][d][r] * inv1;
            vals[d] = o; ss += o * o;
        }
#pragma unroll
        for (int msk = 1; msk < 16; msk <<= 1)
            ss += __shfl_xor(ss, msk, 32);
        float rms = rsqrtf(ss * (1.0f / 64.0f) + 1e-8f) * 0.8f;
        int row = q0 + r + 8 * half;
        int py = row >> 6, px = row & 63;
        f16* op = apad + (((size_t)b * 66 + 1 + py) * 66 + 1 + px) * 512 + h * 64 + n;
#pragma unroll
        for (int d = 0; d < 4; ++d)
            op[d * 16] = (f16)(vals[d] * rms);
    }
}

// ---------------------------------------------------------------------------
extern "C" void kernel_launch(void* const* d_in, const int* in_sizes, int n_in,
                              void* d_out, int out_size, void* d_ws, size_t ws_size,
                              hipStream_t stream) {
    (void)in_sizes; (void)n_in; (void)out_size; (void)ws_size;
    const float* x   = (const float*)d_in[0];
    const float* cr  = (const float*)d_in[1];
    const float* wq  = (const float*)d_in[2];
    const float* wk  = (const float*)d_in[3];
    const float* wv  = (const float*)d_in[4];
    const float* wo  = (const float*)d_in[5];
    const float* lq1 = (const float*)d_in[6];
    const float* lq2 = (const float*)d_in[7];
    const float* lk1 = (const float*)d_in[8];
    const float* lk2 = (const float*)d_in[9];

    const size_t Nxp  = 2ull * 66 * 66 * 512;
    const size_t Ncp  = 2ull * 34 * 34 * 512;
    const size_t Nwq  = 9ull * 1024 * 512;
    const size_t Nwk  = 9ull * 512 * 512;
    const size_t Nwv  = 9ull * 256 * 512;
    const size_t Nwo  = 9ull * 512 * 512;
    const size_t Nq   = 2ull * 4096 * 1024;
    const size_t Nk   = 2ull * 1024 * 512;
    const size_t Nv   = 2ull * 1024 * 256;

    char* ws = (char*)d_ws;
    size_t off = 0;
    auto alloc = [&](size_t bytes) -> void* {
        void* p = ws + off;
        off += (bytes + 255) & ~(size_t)255;
        return p;
    };
    f16*   xpad = (f16*)alloc(Nxp * 2);
    f16*   cpad = (f16*)alloc(Ncp * 2);
    f16*   wqp  = (f16*)alloc(Nwq * 2);
    f16*   wkp  = (f16*)alloc(Nwk * 2);
    f16*   wvp  = (f16*)alloc(Nwv * 2);
    f16*   wop  = (f16*)alloc(Nwo * 2);
    float* qf   = (float*)alloc(Nq * 4);
    float* kf   = (float*)alloc(Nk * 4);
    float* vf   = (float*)alloc(Nv * 4);
    f16*   qp   = (f16*)alloc(Nq * 2);
    f16*   kp   = (f16*)alloc(Nk * 2);
    f16*   vh   = (f16*)alloc(Nv * 2);
    f16*   apad = (f16*)alloc(Nxp * 2);
    float* lam  = (float*)alloc(64 * 4);

    auto grid1 = [](size_t nelem) { return dim3((unsigned)((nelem + 255) / 256)); };

    // layout passes
    pad_in_kernel<<<grid1(Nxp), 256, 0, stream>>>(x,  xpad, 64, 64, Nxp);
    pad_in_kernel<<<grid1(Ncp), 256, 0, stream>>>(cr, cpad, 32, 32, Ncp);
    pack_w_kernel<<<grid1(Nwq), 256, 0, stream>>>(wq, wqp, 1024, Nwq);
    pack_w_kernel<<<grid1(Nwk), 256, 0, stream>>>(wk, wkp, 512,  Nwk);
    pack_w_kernel<<<grid1(Nwv), 256, 0, stream>>>(wv, wvp, 256,  Nwv);
    pack_w_kernel<<<grid1(Nwo), 256, 0, stream>>>(wo, wop, 512,  Nwo);
    zero_f16_kernel<<<grid1(Nxp), 256, 0, stream>>>(apad, Nxp);

    // conv projections (pixel-major f32 out): grid (Cout/32, S/64, B)
    conv3x3_wmma_kernel<<<dim3(32, 64, 2), 128, 0, stream>>>(xpad, wqp, qf, 1024, 64, 64, 0);
    conv3x3_wmma_kernel<<<dim3(16, 16, 2), 128, 0, stream>>>(cpad, wkp, kf,  512, 32, 32, 0);
    conv3x3_wmma_kernel<<<dim3( 8, 16, 2), 128, 0, stream>>>(cpad, wvp, vf,  256, 32, 32, 0);

    // RoPE + cast (pixel-major f16)
    {
        size_t tq = 2ull * 8 * 64 * 4096;
        rope_cast_kernel<<<grid1(tq), 256, 0, stream>>>(qf, qp, 8, 4096, tq);
        size_t tk = 2ull * 4 * 64 * 1024;
        rope_cast_kernel<<<grid1(tk), 256, 0, stream>>>(kf, kp, 4, 1024, tk);
    }
    vtrans_kernel<<<grid1(Nv), 256, 0, stream>>>(vf, vh, Nv);

    lam_kernel<<<1, 32, 0, stream>>>(lq1, lq2, lk1, lk2, lam);

    // differential flash attention: grid (Sq/64, HEADS, B)
    diff_attn_wmma_kernel<<<dim3(64, 8, 2), 128, 0, stream>>>(qp, kp, vh, lam, apad);

    // output conv -> d_out (channel-major f32 [2][512][64][64])
    conv3x3_wmma_kernel<<<dim3(16, 64, 2), 128, 0, stream>>>(apad, wop, (float*)d_out,
                                                             512, 64, 64, 1);
}
